// Feature_Refine_n_35467839930878
// MI455X (gfx1250) — matrix operand order repaired and checked by reading, not compiled, not run on hardware
//
#include <hip/hip_runtime.h>
#include <math.h>

typedef __attribute__((ext_vector_type(16))) _Float16 v16h;
typedef __attribute__((ext_vector_type(8)))  _Float16 v8h;
typedef __attribute__((ext_vector_type(8)))  float    v8f;

#define BB    4
#define HH    64
#define WWI   64
#define LL    4096      /* H*W */
#define TT    16384     /* B*L  */
#define CIN   192
#define NOUT  64
#define DI    384
#define DS    16
#define DTR   12
#define NXP   44        /* DT_RANK + 2*D_STATE */
#define DCONV 4

// ---------------------------------------------------------------- helpers
__device__ __forceinline__ float silu_f(float x) {
  return x / (1.f + __expf(-x));   // x * sigmoid(x)
}

// ---------------------------------------------------------------- f32 -> f16
__global__ void k_f32_to_f16(const float* __restrict__ in, _Float16* __restrict__ out, int n) {
  int i = blockIdx.x * blockDim.x + threadIdx.x;
  if (i < n) out[i] = (_Float16)in[i];
}

// ---------------------------------------------------------------- LayerNorm over C (x is B,C,H*W; output token-major f16)
__global__ void k_layernorm(const float* __restrict__ x, const float* __restrict__ w,
                            const float* __restrict__ bb, _Float16* __restrict__ h) {
  int t = blockIdx.x * blockDim.x + threadIdx.x;
  if (t >= TT) return;
  int b = t >> 12, l = t & (LL - 1);
  const float* xp = x + (size_t)b * CIN * LL + l;
  float s = 0.f, ss = 0.f;
  for (int c = 0; c < CIN; ++c) { float v = xp[(size_t)c * LL]; s += v; ss += v * v; }
  float mu  = s * (1.f / CIN);
  float var = ss * (1.f / CIN) - mu * mu;
  float inv = rsqrtf(var + 1e-5f);
  _Float16* hp = h + (size_t)t * CIN;
  for (int c = 0; c < CIN; ++c) {
    float v = xp[(size_t)c * LL];
    hp[c] = (_Float16)(((v - mu) * inv) * w[c] + bb[c]);
  }
}

// ---------------------------------------------------------------- WMMA GEMM
// C[T x N] = A[T x K](f16, row-major) * W[N x K](f16, row-major)^T (+bias)
// One wave computes a 32(M) x 64(N) strip: 2 A fragments x 4 B fragments ->
// 8 accumulators; every B fragment feeds 2 WMMAs, every A fragment feeds 4.
// 4 waves per block along M. K is a template constant -> K-loop fully unrolls.
//
// Per-lane fragments are CONTIGUOUS in memory (this code's k = grp*16 + s order, same for A and B):
//   A (16x32 f16):  lane = grp*16+m;  halves = A[m][grp*16 .. grp*16+15]
//   B (32x16 f16):  lane = grp*16+n;  halves = W[n][grp*16 .. grp*16+15]
//   C (16x16 f32):  VGPR v : row = m0 + grp*8 + v, col = n0 + (lane&15)
// Out-of-range N lanes clamp their W row to 0 and load unconditionally (their
// D column is never stored), so fragment loads have no divergence.
template <int K>
__global__ void __launch_bounds__(128)
k_wmma_gemm(const _Float16* __restrict__ A, const _Float16* __restrict__ W,
            const float* __restrict__ bias, float* __restrict__ C,
            int T, int N) {
  int wave = threadIdx.x >> 5;
  int lane = threadIdx.x & 31;
  int m0 = (blockIdx.x * 4 + wave) * 32;
  int n0 = blockIdx.y * 64;
  if (m0 >= T) return;
  int grp = lane >> 4;
  int mr  = lane & 15;

  const _Float16* Arow0 = A + (size_t)(m0 + mr) * K + grp * 16;
  const _Float16* Arow1 = Arow0 + (size_t)16 * K;

  int ncol[4];
  const _Float16* Wr[4];
#pragma unroll
  for (int j = 0; j < 4; ++j) {
    int nn = n0 + j * 16 + mr;
    ncol[j] = nn;
    Wr[j] = W + (size_t)(nn < N ? nn : 0) * K + grp * 16;
  }

  v8f acc[2][4];
#pragma unroll
  for (int mi = 0; mi < 2; ++mi)
#pragma unroll
    for (int j = 0; j < 4; ++j) acc[mi][j] = (v8f){};

#pragma unroll
  for (int k0 = 0; k0 < K; k0 += 32) {
    v8h a0lo = *(const v8h*)(Arow0 + k0);
    v8h a0hi = *(const v8h*)(Arow0 + k0 + 8);
    v8h a1lo = *(const v8h*)(Arow1 + k0);
    v8h a1hi = *(const v8h*)(Arow1 + k0 + 8);
    v16h a0 = __builtin_shufflevector(a0lo, a0hi, 0, 1, 2, 3, 4, 5, 6, 7,
                                      8, 9, 10, 11, 12, 13, 14, 15);
    v16h a1 = __builtin_shufflevector(a1lo, a1hi, 0, 1, 2, 3, 4, 5, 6, 7,
                                      8, 9, 10, 11, 12, 13, 14, 15);
#pragma unroll
    for (int j = 0; j < 4; ++j) {
      v16h b = *(const v16h*)(Wr[j] + k0);
      acc[0][j] = __builtin_amdgcn_wmma_f32_16x16x32_f16(false, a0, false, b, (short)0,
                                                         acc[0][j], false, false);
      acc[1][j] = __builtin_amdgcn_wmma_f32_16x16x32_f16(false, a1, false, b, (short)0,
                                                         acc[1][j], false, false);
    }
  }

#pragma unroll
  for (int mi = 0; mi < 2; ++mi) {
    int row0 = m0 + mi * 16 + grp * 8;
#pragma unroll
    for (int j = 0; j < 4; ++j) {
      int col = ncol[j];
      if (col < N) {
        float bv = bias ? bias[col] : 0.0f;
        v8f av = acc[mi][j];
#pragma unroll
        for (int v = 0; v < 8; ++v)
          C[(size_t)(row0 + v) * N + col] = av[v] + bv;
      }
    }
  }
}

// ---------------------------------------------------------------- causal depthwise conv1d + bias + SiLU
__global__ void k_conv1d_silu(const float* __restrict__ xz, const float* __restrict__ cw,
                              const float* __restrict__ cb, float* __restrict__ u32,
                              _Float16* __restrict__ u16) {
  int i = blockIdx.x * blockDim.x + threadIdx.x;
  if (i >= TT * DI) return;
  int t = i / DI, d = i - t * DI;
  int b = t >> 12, l = t & (LL - 1);
  float s = cb[d];
#pragma unroll
  for (int j = 0; j < DCONV; ++j) {
    int ll = l - (DCONV - 1) + j;
    if (ll >= 0)
      s += cw[d * DCONV + j] * xz[(size_t)((b << 12) + ll) * (2 * DI) + d];
  }
  float uu = silu_f(s);
  u32[i] = uu;
  u16[i] = (_Float16)uu;
}

// ---------------------------------------------------------------- dt = softplus(dbl[:,:12] @ dt_proj^T + b)
__global__ void k_dt(const float* __restrict__ dbl, const float* __restrict__ w,
                     const float* __restrict__ bb, float* __restrict__ dt) {
  int i = blockIdx.x * blockDim.x + threadIdx.x;
  if (i >= TT * DI) return;
  int t = i / DI, d = i - t * DI;
  const float* r = dbl + (size_t)t * NXP;
  float s = bb[d];
#pragma unroll
  for (int k = 0; k < DTR; ++k) s += r[k] * w[d * DTR + k];
  dt[i] = (s > 20.f) ? s : log1pf(__expf(s));
}

// ---------------------------------------------------------------- selective scan: thread per (b,d), 16 states in regs
__global__ void k_scan(const float* __restrict__ dt, const float* __restrict__ u,
                       const float* __restrict__ dbl, const float* __restrict__ Alog,
                       float* __restrict__ ys) {
  int i = blockIdx.x * blockDim.x + threadIdx.x;
  if (i >= BB * DI) return;
  int b = i / DI, d = i - b * DI;
  float a[DS], h[DS];
#pragma unroll
  for (int s = 0; s < DS; ++s) { a[s] = -__expf(Alog[d * DS + s]); h[s] = 0.f; }
  for (int l = 0; l < LL; ++l) {
    int t = (b << 12) + l;
    const float* r = dbl + (size_t)t * NXP;
    __builtin_prefetch(r + NXP, 0, 1);   // next token's B/C row -> global_prefetch
    float dtv = dt[(size_t)t * DI + d];
    float du  = dtv * u[(size_t)t * DI + d];
    float acc = 0.f;
#pragma unroll
    for (int s = 0; s < DS; ++s) {
      h[s] = h[s] * __expf(dtv * a[s]) + du * r[DTR + s];
      acc += h[s] * r[DTR + DS + s];
    }
    ys[(size_t)t * DI + d] = acc;
  }
}

// ---------------------------------------------------------------- y = (ys + u*Dp) * silu(z)  -> f16
__global__ void k_gate_y(const float* __restrict__ ys, const float* __restrict__ u,
                         const float* __restrict__ Dp, const float* __restrict__ xz,
                         _Float16* __restrict__ y16) {
  int i = blockIdx.x * blockDim.x + threadIdx.x;
  if (i >= TT * DI) return;
  int t = i / DI, d = i - t * DI;
  float z = xz[(size_t)t * (2 * DI) + DI + d];
  float y = (ys[i] + u[i] * Dp[d]) * silu_f(z);
  y16[i] = (_Float16)y;
}

// ---------------------------------------------------------------- per-(b,c) spatial mean over H*W (LDS reduction)
__global__ void k_spatial_mean(const float* __restrict__ t, float* __restrict__ means) {
  __shared__ float sm[256];
  int bc = blockIdx.x;                 // 0..BB*CIN-1
  int b = bc / CIN, c = bc - b * CIN;
  float s = 0.f;
  for (int l = threadIdx.x; l < LL; l += 256)
    s += t[(size_t)((b << 12) + l) * CIN + c];
  sm[threadIdx.x] = s;
  __syncthreads();
  for (int st = 128; st > 0; st >>= 1) {
    if ((int)threadIdx.x < st) sm[threadIdx.x] += sm[threadIdx.x + st];
    __syncthreads();
  }
  if (threadIdx.x == 0) means[bc] = sm[0] * (1.f / LL);
}

// ---------------------------------------------------------------- t = silu(t + sigma*(t-mean)); in-place + f16 copy
__global__ void k_statnorm_silu(float* __restrict__ t, const float* __restrict__ sigma,
                                const float* __restrict__ means, _Float16* __restrict__ t16) {
  int i = blockIdx.x * blockDim.x + threadIdx.x;
  if (i >= TT * CIN) return;
  int tok = i / CIN, c = i - tok * CIN;
  int b = tok >> 12;
  float v = t[i];
  v = v + sigma[c] * (v - means[b * CIN + c]);
  float w = silu_f(v);
  t[i] = w;
  t16[i] = (_Float16)w;
}

// ---------------------------------------------------------------- generic depthwise 2D conv (input stride CIN)
__global__ void k_dwconv(const float* __restrict__ in, const float* __restrict__ w,
                         const float* __restrict__ bias, float* __restrict__ out,
                         int cin_off, int Cc, int ksz, int dil, int pad) {
  int i = blockIdx.x * blockDim.x + threadIdx.x;
  if (i >= TT * Cc) return;
  int tok = i / Cc, c = i - tok * Cc;
  int b = tok >> 12, hw = tok & (LL - 1);
  int hy = hw >> 6, wx = hw & 63;
  float s = bias[c];
  const float* wc = w + c * ksz * ksz;
  for (int ii = 0; ii < ksz; ++ii) {
    int yy = hy + ii * dil - pad;
    if (yy < 0 || yy >= HH) continue;
    for (int jj = 0; jj < ksz; ++jj) {
      int xx = wx + jj * dil - pad;
      if (xx < 0 || xx >= WWI) continue;
      s += wc[ii * ksz + jj] *
           in[(size_t)((b << 12) + (yy << 6) + xx) * CIN + cin_off + c];
    }
  }
  out[i] = s;
}

// ---------------------------------------------------------------- concat [v0[:,:24], v1, v2] -> f16
__global__ void k_concat_f16(const float* __restrict__ v0, const float* __restrict__ v1,
                             const float* __restrict__ v2, _Float16* __restrict__ out) {
  int i = blockIdx.x * blockDim.x + threadIdx.x;
  if (i >= TT * CIN) return;
  int t = i / CIN, c = i - t * CIN;
  float val;
  if (c < 24)       val = v0[(size_t)t * CIN + c];
  else if (c < 96)  val = v1[(size_t)t * 72 + (c - 24)];
  else              val = v2[(size_t)t * 96 + (c - 96)];
  out[i] = (_Float16)val;
}

// ---------------------------------------------------------------- silu(g)*silu(v) -> f16
__global__ void k_silu_mul_f16(const float* __restrict__ g, const float* __restrict__ v,
                               _Float16* __restrict__ out, int n) {
  int i = blockIdx.x * blockDim.x + threadIdx.x;
  if (i >= n) return;
  out[i] = (_Float16)(silu_f(g[i]) * silu_f(v[i]));
}

// ---------------------------------------------------------------- relu(res + shortcut) -> f16
__global__ void k_add_relu_f16(const float* __restrict__ res, const float* __restrict__ sc,
                               _Float16* __restrict__ out, int n) {
  int i = blockIdx.x * blockDim.x + threadIdx.x;
  if (i >= n) return;
  float v = res[i] + sc[i];
  out[i] = (_Float16)(v > 0.f ? v : 0.f);
}

// ---------------------------------------------------------------- token-major (T x 64) -> (B, 64, H*W)
__global__ void k_write_out(const float* __restrict__ hb, float* __restrict__ out) {
  int i = blockIdx.x * blockDim.x + threadIdx.x;
  if (i >= TT * NOUT) return;
  int t = i / NOUT, o = i - t * NOUT;
  int b = t >> 12, l = t & (LL - 1);
  out[(size_t)((b * NOUT + o) << 12) + l] = hb[i];
}

// ================================================================ launch
extern "C" void kernel_launch(void* const* d_in, const int* in_sizes, int n_in,
                              void* d_out, int out_size, void* d_ws, size_t ws_size,
                              hipStream_t stream) {
  (void)in_sizes; (void)n_in; (void)out_size; (void)ws_size;
  const float* x         = (const float*)d_in[0];
  const float* norm_w    = (const float*)d_in[1];
  const float* norm_b    = (const float*)d_in[2];
  const float* in_proj_w = (const float*)d_in[3];
  const float* conv1d_w  = (const float*)d_in[4];
  const float* conv1d_b  = (const float*)d_in[5];
  const float* x_proj_w  = (const float*)d_in[6];
  const float* dt_proj_w = (const float*)d_in[7];
  const float* dt_proj_b = (const float*)d_in[8];
  const float* A_log     = (const float*)d_in[9];
  const float* Dp        = (const float*)d_in[10];
  const float* out_proj_w= (const float*)d_in[11];
  const float* proj1_w   = (const float*)d_in[12];
  const float* proj1_b   = (const float*)d_in[13];
  const float* sigma     = (const float*)d_in[14];
  const float* gate_w    = (const float*)d_in[15];
  const float* gate_b    = (const float*)d_in[16];
  const float* dw0_w     = (const float*)d_in[17];
  const float* dw0_b     = (const float*)d_in[18];
  const float* dw1_w     = (const float*)d_in[19];
  const float* dw1_b     = (const float*)d_in[20];
  const float* dw2_w     = (const float*)d_in[21];
  const float* dw2_b     = (const float*)d_in[22];
  const float* pw_w      = (const float*)d_in[23];
  const float* pw_b      = (const float*)d_in[24];
  const float* proj2_w   = (const float*)d_in[25];
  const float* proj2_b   = (const float*)d_in[26];
  const float* head_w    = (const float*)d_in[27];
  const float* head_b    = (const float*)d_in[28];
  float* out = (float*)d_out;

  char* ws = (char*)d_ws;
  size_t off = 0;
  auto alloc = [&](size_t bytes) -> void* {
    void* p = ws + off;
    off = (off + bytes + 255) & ~(size_t)255;
    return p;
  };

  // persistent region
  _Float16* w_inproj  = (_Float16*)alloc((size_t)2 * DI * CIN * 2);
  _Float16* w_xproj   = (_Float16*)alloc((size_t)NXP * DI * 2);
  _Float16* w_outproj = (_Float16*)alloc((size_t)CIN * DI * 2);
  _Float16* w_proj1   = (_Float16*)alloc((size_t)CIN * CIN * 2);
  _Float16* w_gate    = (_Float16*)alloc((size_t)CIN * CIN * 2);
  _Float16* w_pw      = (_Float16*)alloc((size_t)CIN * CIN * 2);
  _Float16* w_proj2   = (_Float16*)alloc((size_t)CIN * CIN * 2);
  _Float16* w_head    = (_Float16*)alloc((size_t)NOUT * CIN * 2);
  float*    mamba_out = (float*)alloc((size_t)TT * CIN * 4);
  _Float16* mo16      = (_Float16*)alloc((size_t)TT * CIN * 2);
  size_t stage = off;

  // stage-1 scratch (mamba)
  _Float16* h16 = (_Float16*)alloc((size_t)TT * CIN * 2);
  float*    xz  = (float*)alloc((size_t)TT * 2 * DI * 4);
  float*    u32 = (float*)alloc((size_t)TT * DI * 4);
  _Float16* u16 = (_Float16*)alloc((size_t)TT * DI * 2);
  float*    dbl = (float*)alloc((size_t)TT * NXP * 4);
  float*    dtb = (float*)alloc((size_t)TT * DI * 4);
  float*    ysb = (float*)alloc((size_t)TT * DI * 4);
  _Float16* y16 = (_Float16*)alloc((size_t)TT * DI * 2);

  // stage-2 scratch aliases stage-1 (stage-1 dead after out_proj + convert)
  off = stage;
  float*    tb    = (float*)alloc((size_t)TT * CIN * 4);
  float*    means = (float*)alloc((size_t)BB * CIN * 4);
  _Float16* t16   = (_Float16*)alloc((size_t)TT * CIN * 2);
  float*    gb    = (float*)alloc((size_t)TT * CIN * 4);
  float*    v0    = (float*)alloc((size_t)TT * CIN * 4);
  float*    v1    = (float*)alloc((size_t)TT * 72 * 4);
  float*    v2    = (float*)alloc((size_t)TT * 96 * 4);
  _Float16* vc16  = (_Float16*)alloc((size_t)TT * CIN * 2);
  float*    vb    = (float*)alloc((size_t)TT * CIN * 4);
  _Float16* act16 = (_Float16*)alloc((size_t)TT * CIN * 2);
  float*    resb  = (float*)alloc((size_t)TT * CIN * 4);
  _Float16* res16 = (_Float16*)alloc((size_t)TT * CIN * 2);
  float*    hb    = (float*)alloc((size_t)TT * NOUT * 4);

  const int TPB = 256;
  auto nb = [](long n, int tpb) { return (unsigned)((n + tpb - 1) / tpb); };
  auto ggrid = [](int N) { return dim3(TT / 128, (unsigned)((N + 63) / 64)); };
  dim3 gblk(128);

  // weight conversions
  k_f32_to_f16<<<nb(2 * DI * CIN, TPB), TPB, 0, stream>>>(in_proj_w, w_inproj, 2 * DI * CIN);
  k_f32_to_f16<<<nb(NXP * DI, TPB), TPB, 0, stream>>>(x_proj_w, w_xproj, NXP * DI);
  k_f32_to_f16<<<nb(CIN * DI, TPB), TPB, 0, stream>>>(out_proj_w, w_outproj, CIN * DI);
  k_f32_to_f16<<<nb(CIN * CIN, TPB), TPB, 0, stream>>>(proj1_w, w_proj1, CIN * CIN);
  k_f32_to_f16<<<nb(CIN * CIN, TPB), TPB, 0, stream>>>(gate_w, w_gate, CIN * CIN);
  k_f32_to_f16<<<nb(CIN * CIN, TPB), TPB, 0, stream>>>(pw_w, w_pw, CIN * CIN);
  k_f32_to_f16<<<nb(CIN * CIN, TPB), TPB, 0, stream>>>(proj2_w, w_proj2, CIN * CIN);
  k_f32_to_f16<<<nb(NOUT * CIN, TPB), TPB, 0, stream>>>(head_w, w_head, NOUT * CIN);

  // ---- Mamba stage
  k_layernorm<<<nb(TT, 64), 64, 0, stream>>>(x, norm_w, norm_b, h16);
  k_wmma_gemm<CIN><<<ggrid(2 * DI), gblk, 0, stream>>>(h16, w_inproj, nullptr, xz, TT, 2 * DI);
  k_conv1d_silu<<<nb((long)TT * DI, TPB), TPB, 0, stream>>>(xz, conv1d_w, conv1d_b, u32, u16);
  k_wmma_gemm<DI><<<ggrid(NXP), gblk, 0, stream>>>(u16, w_xproj, nullptr, dbl, TT, NXP);
  k_dt<<<nb((long)TT * DI, TPB), TPB, 0, stream>>>(dbl, dt_proj_w, dt_proj_b, dtb);
  k_scan<<<nb(BB * DI, 128), 128, 0, stream>>>(dtb, u32, dbl, A_log, ysb);
  k_gate_y<<<nb((long)TT * DI, TPB), TPB, 0, stream>>>(ysb, u32, Dp, xz, y16);
  k_wmma_gemm<DI><<<ggrid(CIN), gblk, 0, stream>>>(y16, w_outproj, nullptr, mamba_out, TT, CIN);
  k_f32_to_f16<<<nb((long)TT * CIN, TPB), TPB, 0, stream>>>(mamba_out, mo16, TT * CIN);

  // ---- conv/gate stage
  k_wmma_gemm<CIN><<<ggrid(CIN), gblk, 0, stream>>>(mo16, w_proj1, proj1_b, tb, TT, CIN);
  k_spatial_mean<<<BB * CIN, 256, 0, stream>>>(tb, means);
  k_statnorm_silu<<<nb((long)TT * CIN, TPB), TPB, 0, stream>>>(tb, sigma, means, t16);
  k_wmma_gemm<CIN><<<ggrid(CIN), gblk, 0, stream>>>(t16, w_gate, gate_b, gb, TT, CIN);
  k_dwconv<<<nb((long)TT * CIN, TPB), TPB, 0, stream>>>(tb, dw0_w, dw0_b, v0, 0, CIN, 5, 1, 2);
  k_dwconv<<<nb((long)TT * 72, TPB), TPB, 0, stream>>>(v0, dw1_w, dw1_b, v1, 24, 72, 5, 2, 4);
  k_dwconv<<<nb((long)TT * 96, TPB), TPB, 0, stream>>>(v0, dw2_w, dw2_b, v2, 96, 96, 7, 3, 9);
  k_concat_f16<<<nb((long)TT * CIN, TPB), TPB, 0, stream>>>(v0, v1, v2, vc16);
  k_wmma_gemm<CIN><<<ggrid(CIN), gblk, 0, stream>>>(vc16, w_pw, pw_b, vb, TT, CIN);
  k_silu_mul_f16<<<nb((long)TT * CIN, TPB), TPB, 0, stream>>>(gb, vb, act16, TT * CIN);
  k_wmma_gemm<CIN><<<ggrid(CIN), gblk, 0, stream>>>(act16, w_proj2, proj2_b, resb, TT, CIN);
  k_add_relu_f16<<<nb((long)TT * CIN, TPB), TPB, 0, stream>>>(resb, mamba_out, res16, TT * CIN);
  k_wmma_gemm<CIN><<<ggrid(NOUT), gblk, 0, stream>>>(res16, w_head, head_b, hb, TT, NOUT);
  k_write_out<<<nb((long)TT * NOUT, TPB), TPB, 0, stream>>>(hb, out);
}
